// SAModule_62878321213704
// MI455X (gfx1250) — compile-verified, matrix operand order checked by reading, not run on hardware
//
#include <hip/hip_runtime.h>
#include <hip/hip_bf16.h>

// ---------------- static problem config (matches reference) ----------------
#define BN   16          // clouds
#define NP   4096        // points per cloud
#define SC   1024        // centroids per cloud (ratio 0.25)
#define KN   64          // max neighbors
#define FIN  64          // input feature dim
#define R2C  0.0225f     // R^2, R=0.15
#define CURW 10.0f       // curvature scalar

typedef __attribute__((ext_vector_type(16))) _Float16 v16h;
typedef __attribute__((ext_vector_type(8)))  _Float16 v8h;
typedef __attribute__((ext_vector_type(8)))  float    v8f;

union V16H { v16h v; v8h h[2]; };

// ---------------------------------------------------------------------------
// A-fragment (16x32 f16, M x K) from a row-major f16 LDS tile.
// ISA 7.12.2: lane L (g=L/16, r=L%16) holds row M=r; elements e=0..7 are
// K = kbase + g*8 + e, elements e=8..15 are K = kbase + 16 + g*8 + (e-8).
// Both runs are 8 contiguous halfs => two 16-byte LDS loads.
// ---------------------------------------------------------------------------
__device__ __forceinline__ v16h lds_a_frag(const _Float16* base, int rowStride,
                                           int row0, int kbase, int lane) {
    const int g = lane >> 4, r = lane & 15;
    const _Float16* p = base + (row0 + r) * rowStride + kbase + g * 8;
    V16H u;
    u.h[0] = *(const v8h*)(p);
    u.h[1] = *(const v8h*)(p + 16);
    return u.v;
}

// B-fragment from pre-shuffled global weights: frag f, lane L -> 16 contiguous
// halfs at wsh[(f*32+L)*16]  (two global_load_b128).
__device__ __forceinline__ v16h ld_b_frag(const _Float16* wsh, int frag, int lane) {
    const _Float16* p = wsh + (((size_t)frag * 32 + lane) << 4);
    V16H u;
    u.h[0] = *(const v8h*)(p);
    u.h[1] = *(const v8h*)(p + 8);
    return u.v;
}

// ---------------------------------------------------------------------------
// Kernel 0: repack fp32 weights into f16 WMMA B-fragment layout.
// W1: [67x64] pad K->96 : frags 0..11   (kc 0..2, nt 0..3)
// W2: [64x64]           : frags 12..19  (kc 0..1, nt 0..3)
// W3: [64x128]          : frags 20..35  (kc 0..1, nt 0..7)
// ---------------------------------------------------------------------------
__global__ void wshuffle_kernel(const float* __restrict__ W1,
                                const float* __restrict__ W2,
                                const float* __restrict__ W3,
                                _Float16* __restrict__ wsh) {
    const int f = blockIdx.x;        // 0..35
    const int lane = threadIdx.x;    // 0..31
    const int g = lane >> 4, n0 = lane & 15;
    const float* W; int Kreal, Nw, kc, nt;
    if (f < 12)       { W = W1; Kreal = 67; Nw = 64;  kc = f >> 2;       nt = f & 3; }
    else if (f < 20)  { W = W2; Kreal = 64; Nw = 64;  kc = (f-12) >> 2;  nt = (f-12) & 3; }
    else              { W = W3; Kreal = 64; Nw = 128; kc = (f-20) >> 3;  nt = (f-20) & 7; }
    _Float16* dst = wsh + (((size_t)f * 32 + lane) << 4);
    const int n = nt * 16 + n0;
    #pragma unroll
    for (int e = 0; e < 16; ++e) {
        const int kl = (e < 8) ? (g * 8 + e) : (16 + g * 8 + (e - 8));
        const int k  = kc * 32 + kl;
        const float v = (k < Kreal) ? W[(size_t)k * Nw + n] : 0.0f;
        dst[e] = (_Float16)v;
    }
}

// ---------------------------------------------------------------------------
// Kernel 1: curvature-weighted FPS. One 1024-thread block per cloud;
// each thread owns 4 points (strided). pos staged in LDS (48 KB).
// Tie-break = smaller index (matches jnp.argmax "first max").
// ---------------------------------------------------------------------------
__global__ __launch_bounds__(1024) void fps_kernel(const float* __restrict__ pos,
                                                   const float* __restrict__ curv,
                                                   int* __restrict__ idx_out) {
    __shared__ float xs[NP], ys[NP], zs[NP];
    __shared__ float wval[32];
    __shared__ int   widx[32];
    __shared__ float selx, sely, selz;

    const int b = blockIdx.x;
    const int t = threadIdx.x;
    const int lane = t & 31, wave = t >> 5;
    const float* pb = pos + (size_t)b * NP * 3;

    float px[4], py[4], pz[4], wgt[4], mind[4];
    #pragma unroll
    for (int u = 0; u < 4; ++u) {
        const int i = t + u * 1024;
        const float X = pb[i * 3 + 0], Y = pb[i * 3 + 1], Z = pb[i * 3 + 2];
        xs[i] = X; ys[i] = Y; zs[i] = Z;
        px[u] = X; py[u] = Y; pz[u] = Z;
        wgt[u] = 1.0f + CURW * curv[(size_t)b * NP + i];
    }
    __syncthreads();
    const float x0 = xs[0], y0 = ys[0], z0 = zs[0];
    #pragma unroll
    for (int u = 0; u < 4; ++u) {
        const float dx = px[u] - x0, dy = py[u] - y0, dz = pz[u] - z0;
        mind[u] = dx * dx + dy * dy + dz * dz;
    }
    if (t == 0) idx_out[b * SC + 0] = 0;

    for (int s = 1; s < SC; ++s) {
        float best = -1.0f; int bi = 0x7fffffff;
        #pragma unroll
        for (int u = 0; u < 4; ++u) {
            const float sc = mind[u] * wgt[u];
            if (sc > best) { best = sc; bi = t + u * 1024; }
        }
        #pragma unroll
        for (int off = 16; off >= 1; off >>= 1) {
            const float ov = __shfl_down(best, off, 32);
            const int   oi = __shfl_down(bi, off, 32);
            if (ov > best || (ov == best && oi < bi)) { best = ov; bi = oi; }
        }
        if (lane == 0) { wval[wave] = best; widx[wave] = bi; }
        __syncthreads();
        if (wave == 0) {
            best = wval[lane]; bi = widx[lane];
            #pragma unroll
            for (int off = 16; off >= 1; off >>= 1) {
                const float ov = __shfl_down(best, off, 32);
                const int   oi = __shfl_down(bi, off, 32);
                if (ov > best || (ov == best && oi < bi)) { best = ov; bi = oi; }
            }
            if (lane == 0) {
                selx = xs[bi]; sely = ys[bi]; selz = zs[bi];
                idx_out[b * SC + s] = bi;
            }
        }
        __syncthreads();
        const float cxx = selx, cyy = sely, czz = selz;
        #pragma unroll
        for (int u = 0; u < 4; ++u) {
            const float dx = px[u] - cxx, dy = py[u] - cyy, dz = pz[u] - czz;
            mind[u] = fminf(mind[u], dx * dx + dy * dy + dz * dz);
        }
    }
}

// ---------------------------------------------------------------------------
// Kernel 2: ball query, exact K=64 nearest within R via 4-pass radix select
// on float bits of d^2 (monotone unsigned for d^2 >= 0). One 256-thread WG
// per centroid; 16 points per thread kept in registers.
// ---------------------------------------------------------------------------
__global__ __launch_bounds__(256) void ballq_kernel(const float* __restrict__ pos,
                                                    const int* __restrict__ idxbuf,
                                                    int* __restrict__ cntbuf,
                                                    int* __restrict__ nbrbuf) {
    __shared__ int hist[256];
    __shared__ int sC, sNeed, cLt, cEq;
    __shared__ unsigned int sPrefix;

    const int blk = blockIdx.x;          // centroid id
    const int b = blk >> 10;
    const int t = threadIdx.x;
    const int ci = idxbuf[blk];
    const float* pb = pos + (size_t)b * NP * 3;
    const float cx = pb[ci * 3 + 0], cy = pb[ci * 3 + 1], cz = pb[ci * 3 + 2];

    if (t == 0) { sC = 0; cLt = 0; cEq = 0; }
    __syncthreads();

    unsigned int ub[16];
    int mycount = 0;
    #pragma unroll
    for (int u = 0; u < 16; ++u) {
        const int i = t + u * 256;
        const float dx = pb[i * 3 + 0] - cx, dy = pb[i * 3 + 1] - cy, dz = pb[i * 3 + 2] - cz;
        const float d2 = dx * dx + dy * dy + dz * dz;
        unsigned int v = 0xFFFFFFFFu;
        if (d2 <= R2C) { v = __float_as_uint(d2); ++mycount; }
        ub[u] = v;
    }
    atomicAdd(&sC, mycount);
    __syncthreads();
    const int C = sC;
    int* row = nbrbuf + (size_t)blk * KN;

    if (C <= KN) {
        #pragma unroll
        for (int u = 0; u < 16; ++u)
            if (ub[u] != 0xFFFFFFFFu) {
                const int slot = atomicAdd(&cLt, 1);
                row[slot] = b * NP + (t + u * 256);
            }
        if (t == 0) cntbuf[blk] = C;
    } else {
        unsigned int prefix = 0, pmask = 0;
        int need = KN;                   // 1-indexed rank within filtered subset
        for (int pass = 0; pass < 4; ++pass) {
            const int sh = 24 - 8 * pass;
            hist[t] = 0;
            __syncthreads();
            #pragma unroll
            for (int u = 0; u < 16; ++u) {
                const unsigned int v = ub[u];
                if ((v & pmask) == prefix) atomicAdd(&hist[(v >> sh) & 255], 1);
            }
            __syncthreads();
            if (t == 0) {
                int cum = 0, sel = 255, rem = need;
                for (int q = 0; q < 256; ++q) {
                    const int h = hist[q];
                    if (cum + h >= need) { sel = q; rem = need - cum; break; }
                    cum += h;
                }
                sNeed = rem;
                sPrefix = prefix | ((unsigned int)sel << sh);
            }
            __syncthreads();
            prefix = sPrefix; need = sNeed;
            pmask |= (0xFFu << sh);
        }
        // prefix == bits of rank-64 d^2; need = #ties to keep; L = #strictly-less
        const int L = KN - need;
        #pragma unroll
        for (int u = 0; u < 16; ++u) {
            const unsigned int v = ub[u];
            const int gi = b * NP + (t + u * 256);
            if (v < prefix) {
                const int slot = atomicAdd(&cLt, 1);
                row[slot] = gi;
            } else if (v == prefix) {
                const int e = atomicAdd(&cEq, 1);
                if (e < need) row[L + e] = gi;
            }
        }
        if (t == 0) cntbuf[blk] = KN;
    }
}

// ---------------------------------------------------------------------------
// Kernel 3: PointNetConv MLP via WMMA. One 4-wave WG per centroid.
// Wave w owns M-tile w (neighbor rows 16w..16w+15) for all three layers,
// so only the gather->L1 and pmax barriers are strictly required.
// ---------------------------------------------------------------------------
__global__ __launch_bounds__(128) void mlp_kernel(const float* __restrict__ x,
                                                  const float* __restrict__ pos,
                                                  const float* __restrict__ b1,
                                                  const float* __restrict__ b2,
                                                  const float* __restrict__ b3,
                                                  const int* __restrict__ idxbuf,
                                                  const int* __restrict__ cntbuf,
                                                  const int* __restrict__ nbrbuf,
                                                  const _Float16* __restrict__ wsh,
                                                  float* __restrict__ out) {
    __shared__ _Float16 h0[64 * 96];     // [neighbor][feat: x(64)|rel(3)|pad(29)]
    __shared__ _Float16 h1[64 * 64];
    __shared__ _Float16 h2[64 * 64];
    __shared__ float    pmax[4][128];

    const int blk = blockIdx.x;
    const int b = blk >> 10;
    const int t = threadIdx.x;
    const int lane = t & 31, wave = t >> 5;
    const int g = lane >> 4, n0 = lane & 15;

    const int cnt = cntbuf[blk];
    const int ci = idxbuf[blk];
    const size_t cg = (size_t)b * NP + ci;
    const float cx = pos[cg * 3 + 0], cy = pos[cg * 3 + 1], cz = pos[cg * 3 + 2];

    // ---- gather neighbor features into h0 (f16, zero-padded) ----
    {
        const int j = t >> 1, half = t & 1;
        const int jg = (j < cnt) ? nbrbuf[(size_t)blk * KN + j] : -1;
        _Float16* row = h0 + j * 96;
        if (half == 0) {
            if (jg >= 0) {
                const float4* xp = (const float4*)(x + (size_t)jg * FIN);
                #pragma unroll
                for (int q = 0; q < 12; ++q) {
                    const float4 v4 = xp[q];
                    row[q*4+0] = (_Float16)v4.x; row[q*4+1] = (_Float16)v4.y;
                    row[q*4+2] = (_Float16)v4.z; row[q*4+3] = (_Float16)v4.w;
                }
            } else {
                #pragma unroll
                for (int c = 0; c < 48; ++c) row[c] = (_Float16)0.0f;
            }
        } else {
            if (jg >= 0) {
                const float4* xp = (const float4*)(x + (size_t)jg * FIN);
                #pragma unroll
                for (int q = 12; q < 16; ++q) {
                    const float4 v4 = xp[q];
                    row[q*4+0] = (_Float16)v4.x; row[q*4+1] = (_Float16)v4.y;
                    row[q*4+2] = (_Float16)v4.z; row[q*4+3] = (_Float16)v4.w;
                }
                row[64] = (_Float16)(pos[(size_t)jg * 3 + 0] - cx);
                row[65] = (_Float16)(pos[(size_t)jg * 3 + 1] - cy);
                row[66] = (_Float16)(pos[(size_t)jg * 3 + 2] - cz);
            } else {
                #pragma unroll
                for (int c = 48; c < 67; ++c) row[c] = (_Float16)0.0f;
            }
            #pragma unroll
            for (int c = 67; c < 96; ++c) row[c] = (_Float16)0.0f;
        }
    }
    __syncthreads();

    const _Float16* w1 = wsh;
    const _Float16* w2 = wsh + 12 * 512;
    const _Float16* w3 = wsh + 20 * 512;
    const int mt = wave;

    // ---- layer 1: [16x96] @ [96x64], bias+ReLU -> h1 ----
    #pragma unroll
    for (int nt = 0; nt < 4; ++nt) {
        v8f acc = {};
        #pragma unroll
        for (int kc = 0; kc < 3; ++kc) {
            const v16h a  = lds_a_frag(h0, 96, mt * 16, kc * 32, lane);
            const v16h bf = ld_b_frag(w1, kc * 4 + nt, lane);
            acc = __builtin_amdgcn_wmma_f32_16x16x32_f16(false, a, false, bf,
                                                         (short)0, acc, false, false);
        }
        const float bias = b1[nt * 16 + n0];
        #pragma unroll
        for (int v = 0; v < 8; ++v) {
            float val = acc[v] + bias;
            val = val > 0.0f ? val : 0.0f;
            h1[(mt * 16 + v + 8 * g) * 64 + nt * 16 + n0] = (_Float16)val;
        }
    }
    __syncthreads();

    // ---- layer 2: [16x64] @ [64x64], bias+ReLU -> h2 ----
    #pragma unroll
    for (int nt = 0; nt < 4; ++nt) {
        v8f acc = {};
        #pragma unroll
        for (int kc = 0; kc < 2; ++kc) {
            const v16h a  = lds_a_frag(h1, 64, mt * 16, kc * 32, lane);
            const v16h bf = ld_b_frag(w2, kc * 4 + nt, lane);
            acc = __builtin_amdgcn_wmma_f32_16x16x32_f16(false, a, false, bf,
                                                         (short)0, acc, false, false);
        }
        const float bias = b2[nt * 16 + n0];
        #pragma unroll
        for (int v = 0; v < 8; ++v) {
            float val = acc[v] + bias;
            val = val > 0.0f ? val : 0.0f;
            h2[(mt * 16 + v + 8 * g) * 64 + nt * 16 + n0] = (_Float16)val;
        }
    }
    __syncthreads();

    // ---- layer 3: [16x64] @ [64x128], bias+ReLU, masked row-max ----
    #pragma unroll
    for (int nt = 0; nt < 8; ++nt) {
        v8f acc = {};
        #pragma unroll
        for (int kc = 0; kc < 2; ++kc) {
            const v16h a  = lds_a_frag(h2, 64, mt * 16, kc * 32, lane);
            const v16h bf = ld_b_frag(w3, kc * 8 + nt, lane);
            acc = __builtin_amdgcn_wmma_f32_16x16x32_f16(false, a, false, bf,
                                                         (short)0, acc, false, false);
        }
        const float bias = b3[nt * 16 + n0];
        float mx = -__builtin_inff();
        #pragma unroll
        for (int v = 0; v < 8; ++v) {
            const int rowm = mt * 16 + v + 8 * g;       // C layout: m = v + 8*(lane/16)
            float val = acc[v] + bias;
            val = val > 0.0f ? val : 0.0f;
            if (rowm < cnt) mx = fmaxf(mx, val);
        }
        mx = fmaxf(mx, __shfl_xor(mx, 16, 32));         // fold rows 8..15 onto 0..7
        if (lane < 16) pmax[wave][nt * 16 + lane] = mx;
    }
    __syncthreads();

    const float mx = fmaxf(fmaxf(pmax[0][t], pmax[1][t]),
                           fmaxf(pmax[2][t], pmax[3][t]));
    out[(size_t)blk * 128 + t] = mx;
}

// ---------------------------------------------------------------------------
// Kernel 4: gather pos / batch / curvature outputs for the sampled set.
// ---------------------------------------------------------------------------
__global__ void tail_kernel(const float* __restrict__ pos,
                            const int* __restrict__ batch,
                            const float* __restrict__ curv,
                            const int* __restrict__ idxbuf,
                            float* __restrict__ out) {
    const int i = blockIdx.x * 256 + threadIdx.x;
    if (i >= BN * SC) return;
    const int b = i >> 10;
    const size_t gidx = (size_t)b * NP + idxbuf[i];
    float* out_pos   = out + (size_t)BN * SC * 128;
    float* out_batch = out_pos + (size_t)BN * SC * 3;
    float* out_curv  = out_batch + (size_t)BN * SC;
    out_pos[(size_t)i * 3 + 0] = pos[gidx * 3 + 0];
    out_pos[(size_t)i * 3 + 1] = pos[gidx * 3 + 1];
    out_pos[(size_t)i * 3 + 2] = pos[gidx * 3 + 2];
    out_batch[i] = (float)batch[gidx];
    out_curv[i]  = curv[gidx];
}

// ---------------------------------------------------------------------------
extern "C" void kernel_launch(void* const* d_in, const int* in_sizes, int n_in,
                              void* d_out, int out_size, void* d_ws, size_t ws_size,
                              hipStream_t stream) {
    (void)in_sizes; (void)n_in; (void)out_size; (void)ws_size;
    const float* x    = (const float*)d_in[0];
    const float* pos  = (const float*)d_in[1];
    const int*   batch= (const int*)  d_in[2];
    const float* curv = (const float*)d_in[3];
    const float* W1   = (const float*)d_in[4];
    const float* b1   = (const float*)d_in[5];
    const float* W2   = (const float*)d_in[6];
    const float* b2   = (const float*)d_in[7];
    const float* W3   = (const float*)d_in[8];
    const float* b3   = (const float*)d_in[9];
    float* out = (float*)d_out;

    char* ws = (char*)d_ws;
    int* idxbuf = (int*)(ws);                                   // 16384 ints
    int* cntbuf = (int*)(ws + 65536);                           // 16384 ints
    int* nbrbuf = (int*)(ws + 131072);                          // 16384*64 ints (4 MB)
    _Float16* wsh = (_Float16*)(ws + 131072 + (size_t)BN * SC * KN * 4); // 36 KB, 16B-aligned

    wshuffle_kernel<<<36, 32, 0, stream>>>(W1, W2, W3, wsh);
    fps_kernel<<<BN, 1024, 0, stream>>>(pos, curv, idxbuf);
    ballq_kernel<<<BN * SC, 256, 0, stream>>>(pos, idxbuf, cntbuf, nbrbuf);
    mlp_kernel<<<BN * SC, 128, 0, stream>>>(x, pos, b1, b2, b3,
                                            idxbuf, cntbuf, nbrbuf, wsh, out);
    tail_kernel<<<(BN * SC + 255) / 256, 256, 0, stream>>>(pos, batch, curv, idxbuf, out);
}
